// PeepholeLSTMCell_8873402434093
// MI455X (gfx1250) — matrix operand order repaired and checked by reading, not compiled, not run on hardware
//
#include <hip/hip_runtime.h>
#include <stdint.h>

// ---- CDNA5 vector types ----
typedef __attribute__((ext_vector_type(16))) __bf16    v16bf;
typedef __attribute__((ext_vector_type(8)))  float     v8f;
typedef __attribute__((ext_vector_type(4)))  uint32_t  u32x4;
typedef __attribute__((ext_vector_type(2)))  uint32_t  u32x2;
typedef __attribute__((ext_vector_type(4)))  float     f32x4;

// explicit global-address-space pointer: forces global_load_* (not flat_load_*)
typedef const float     __attribute__((address_space(1)))* gfp;
typedef const f32x4     __attribute__((address_space(1)))* gfp4;

#define BATCH 8192
#define ISZ   512
#define HSZ   1024
#define KTOT  (ISZ + HSZ)   // 1536
#define BM    64
#define BN    128
#define KT    32
#define LDK   56            // 32 + 24 pad: 112B row stride, 16B aligned, bank-conflict free
#define NT    512           // threads per block (16 waves)

union Frag { u32x4 q[2]; v16bf v; };

__device__ __forceinline__ uint32_t pack2bf(float a, float b) {
  uint32_t r;                       // v_cvt_pk_bf16_f32: HW packed f32->bf16 (RNE)
  asm("v_cvt_pk_bf16_f32 %0, %1, %2" : "=v"(r) : "v"(a), "v"(b));
  return r;
}

__device__ __forceinline__ float sigm(float x)   { return 1.f / (1.f + __expf(-x)); }
__device__ __forceinline__ float tanh_f(float x) { return 2.f / (1.f + __expf(-2.f * x)) - 1.f; }

__global__ __launch_bounds__(NT, 1)
void lstm_wmma_kernel(const float* __restrict__ x,   const float* __restrict__ hp,
                      const float* __restrict__ cp,
                      const float* __restrict__ Wci, const float* __restrict__ Wcf,
                      const float* __restrict__ Wco,
                      const float* __restrict__ Wif, const float* __restrict__ Whf,
                      const float* __restrict__ Wii, const float* __restrict__ Whi,
                      const float* __restrict__ Wig, const float* __restrict__ Whg,
                      const float* __restrict__ Wio, const float* __restrict__ Who,
                      float* __restrict__ out) {
  // ping-pong LDS: 2 x (7.0 + 56.0) KB staged tiles = 126 KB of the 320 KB WGP pool
  __shared__ __attribute__((aligned(16))) uint16_t sA[2][BM][LDK];       // [buf][m][k] bf16
  __shared__ __attribute__((aligned(16))) uint16_t sB[2][4][BN][LDK];    // [buf][gate][n][k] bf16

  const int t      = threadIdx.x;
  const int lane   = t & 31;
  const int wave   = t >> 5;           // 0..15
  const int wave_m = wave & 1;         // 0..1
  const int wave_n = wave >> 1;        // 0..7
  const int rb     = wave_m * 32;      // 32 rows per wave
  const int cb     = wave_n * 16;      // 16 cols per wave
  const int lcol   = lane & 15;
  const int lhalf  = lane >> 4;

  const int m0 = blockIdx.x * BM;
  const int n0 = blockIdx.y * BN;

  // per-thread staging coordinates (one 8k x 4n transpose block per thread)
  const int gth  = t >> 7;             // gate this thread stages: 0..3
  const int kb   = ((t >> 5) & 3) * 8; // k offset within tile
  const int nn   = (t & 31) * 4;       // n offset within tile
  const int arow = t >> 3;             // A-tile row (0..63)
  const int akf  = (t & 7) * 4;        // A-tile k offset

  // loop-invariant staging base pointers (gate select hoisted out of K-loop)
  const gfp Wlo = (gfp)((gth == 0) ? Wif : (gth == 1) ? Wii : (gth == 2) ? Wig : Wio)
                  + (size_t)kb * HSZ + n0 + nn;
  const gfp Whg_ = (gfp)((gth == 0) ? Whf : (gth == 1) ? Whi : (gth == 2) ? Whg : Who)
                  + (size_t)kb * HSZ + n0 + nn;
  const gfp Axp = (gfp)x  + (size_t)(m0 + arow) * ISZ + akf;
  const gfp Ahp = (gfp)hp + (size_t)(m0 + arow) * HSZ + akf;

  v8f acc[4][2];                       // [gate][tm] : 32x16 per gate per wave
  #pragma unroll
  for (int g = 0; g < 4; ++g)
    #pragma unroll
    for (int tm = 0; tm < 2; ++tm)
      acc[g][tm] = (v8f){0.f,0.f,0.f,0.f,0.f,0.f,0.f,0.f};

  // ---- software-pipelined staging buffers ----
  f32x4 rawA;
  f32x4 rawB[8];
  uint32_t apk[2];
  uint32_t bpk[4][4];

  auto issue_loads = [&](int k0) {
    const bool ph = (k0 >= ISZ);
    const int  kr = ph ? (k0 - ISZ) : k0;
    rawA = *(gfp4)((ph ? Ahp : Axp) + kr);
    const gfp p = (ph ? Whg_ : Wlo) + (size_t)kr * HSZ;
    #pragma unroll
    for (int rr = 0; rr < 8; ++rr)
      rawB[rr] = *(gfp4)(p + (size_t)rr * HSZ);
  };

  auto pack_all = [&]() {
    apk[0] = pack2bf(rawA.x, rawA.y);
    apk[1] = pack2bf(rawA.z, rawA.w);
    #pragma unroll
    for (int c = 0; c < 4; ++c) {
      bpk[c][0] = pack2bf(rawB[0][c], rawB[1][c]);
      bpk[c][1] = pack2bf(rawB[2][c], rawB[3][c]);
      bpk[c][2] = pack2bf(rawB[4][c], rawB[5][c]);
      bpk[c][3] = pack2bf(rawB[6][c], rawB[7][c]);
    }
  };

  auto store_lds = [&](uint16_t (*dA)[LDK], uint16_t (*dB)[BN][LDK]) {
    *(u32x2*)&dA[arow][akf] = (u32x2){apk[0], apk[1]};
    #pragma unroll
    for (int c = 0; c < 4; ++c)
      *(u32x4*)&dB[gth][nn + c][kb] = (u32x4){bpk[c][0], bpk[c][1], bpk[c][2], bpk[c][3]};
  };

  // one K-step: barrier, prefetch next, 8 WMMAs from cur buffer, pack+store into nxt buffer
  auto body = [&](int k0, uint16_t (*cA)[LDK], uint16_t (*cB)[BN][LDK],
                          uint16_t (*nA)[LDK], uint16_t (*nB)[BN][LDK]) {
    __syncthreads();                       // cur tiles visible; all readers of nxt done
    const bool more = (k0 + KT) < KTOT;
    if (more) issue_loads(k0 + KT);        // global loads fly under the WMMAs below

    Frag a[2];
    #pragma unroll
    for (int tm = 0; tm < 2; ++tm) {
      const uint16_t* p = &cA[rb + tm * 16 + lcol][lhalf * 8];
      a[tm].q[0] = *(const u32x4*)p;       // K {0..7} / {8..15}
      a[tm].q[1] = *(const u32x4*)(p + 16);// K {16..23} / {24..31}
    }
    Frag b[4];
    #pragma unroll
    for (int g = 0; g < 4; ++g) {
      const uint16_t* p = &cB[g][cb + lcol][lhalf * 8];
      b[g].q[0] = *(const u32x4*)p;        // K {0..7} / {8..15}
      b[g].q[1] = *(const u32x4*)(p + 16); // K {16..23} / {24..31}
    }

    #pragma unroll
    for (int g = 0; g < 4; ++g)
      #pragma unroll
      for (int tm = 0; tm < 2; ++tm)
        acc[g][tm] = __builtin_amdgcn_wmma_f32_16x16x32_bf16(
            false, a[tm].v, false, b[g].v, (short)0, acc[g][tm], false, false);

    if (more) { pack_all(); store_lds(nA, nB); }  // waits for in-flight loads only here
  };

  // ---- prologue: stage K-step 0 into buffer 0 ----
  issue_loads(0);
  pack_all();
  store_lds(sA[0], sB[0]);

  for (int k0 = 0; k0 < KTOT; k0 += 2 * KT) {     // 24 double-steps, buffers alternate
    body(k0,      sA[0], sB[0], sA[1], sB[1]);
    body(k0 + KT, sA[1], sB[1], sA[0], sB[0]);
  }

  // ---- epilogue: peephole gates + activations ----
  const size_t BH = (size_t)BATCH * HSZ;
  float* __restrict__ outH = out;
  float* __restrict__ outC = out + BH;

  const int col = n0 + cb + lcol;
  const float wci = Wci[col];
  const float wcf = Wcf[col];
  const float wco = Wco[col];
  #pragma unroll
  for (int tm = 0; tm < 2; ++tm) {
    #pragma unroll
    for (int r = 0; r < 8; ++r) {
      const int row = m0 + rb + tm * 16 + r + lhalf * 8;
      const size_t off = (size_t)row * HSZ + col;
      const float c  = __builtin_nontemporal_load(cp + off);   // streamed once
      const float fg = sigm(acc[0][tm][r] + c * wcf);
      const float ig = sigm(acc[1][tm][r] + c * wci);
      const float gg = tanh_f(acc[2][tm][r]);
      const float cs = fg * c + ig * gg;
      const float og = sigm(acc[3][tm][r] + cs * wco);
      __builtin_nontemporal_store(og * tanh_f(cs), outH + off);
      __builtin_nontemporal_store(cs, outC + off);
    }
  }
}

extern "C" void kernel_launch(void* const* d_in, const int* in_sizes, int n_in,
                              void* d_out, int out_size, void* d_ws, size_t ws_size,
                              hipStream_t stream) {
  const float* x   = (const float*)d_in[0];
  const float* hp  = (const float*)d_in[1];
  const float* cp  = (const float*)d_in[2];
  const float* Wci = (const float*)d_in[3];
  const float* Wcf = (const float*)d_in[4];
  const float* Wco = (const float*)d_in[5];
  const float* Wif = (const float*)d_in[6];
  const float* Whf = (const float*)d_in[7];
  const float* Wii = (const float*)d_in[8];
  const float* Whi = (const float*)d_in[9];
  const float* Wig = (const float*)d_in[10];
  const float* Whg = (const float*)d_in[11];
  const float* Wio = (const float*)d_in[12];
  const float* Who = (const float*)d_in[13];

  dim3 grid(BATCH / BM, HSZ / BN);
  lstm_wmma_kernel<<<grid, NT, 0, stream>>>(x, hp, cp, Wci, Wcf, Wco,
                                            Wif, Whf, Wii, Whi, Wig, Whg, Wio, Who,
                                            (float*)d_out);
}
